// SelfAttentionUncertainty_58463094833355
// MI455X (gfx1250) — matrix operand order repaired and checked by reading, not compiled, not run on hardware
//
#include <hip/hip_runtime.h>
#include <hip/hip_bf16.h>
#include <stdint.h>

typedef _Float16 h16;
typedef __attribute__((ext_vector_type(16))) _Float16 v16h;
typedef __attribute__((ext_vector_type(8)))  _Float16 v8h;
typedef __attribute__((ext_vector_type(8)))  float    v8f;
typedef __attribute__((ext_vector_type(4)))  int      v4i;

#define BATCH 8
#define CDIM 768
#define NPOS 1024
#define NHEADS 12
#define HDIM 64
#define QKV_ROWS 2304
#define ATTN_SCALE 0.125f               /* 64^-0.5 */
#define LOG10K_OVER_PAIRS 0.57564627324851142f /* ln(10000)/16 */

#if defined(__has_builtin)
#  if __has_builtin(__builtin_amdgcn_global_load_async_to_lds_b128) && \
      __has_builtin(__builtin_amdgcn_s_wait_asynccnt)
#    define HAVE_ASYNC 1
#  endif
#endif
#ifndef HAVE_ASYNC
#define HAVE_ASYNC 0
#endif

typedef __attribute__((address_space(1))) v4i* as1_v4i_p;
typedef __attribute__((address_space(3))) v4i* as3_v4i_p;

// copy 8 halves (16B) global -> LDS (async DMA when available)
__device__ __forceinline__ void cp16(h16* l, const h16* g) {
#if HAVE_ASYNC
  __builtin_amdgcn_global_load_async_to_lds_b128((as1_v4i_p)g, (as3_v4i_p)l, 0, 0);
#else
  *(v8h*)l = *(const v8h*)g;
#endif
}
// wait until at most the newest 4 async ops are outstanding (prev tile done;
// ASYNCcnt completes in order per wave)
__device__ __forceinline__ void cp_wait4() {
#if HAVE_ASYNC
  __builtin_amdgcn_s_wait_asynccnt(4);
#endif
}
__device__ __forceinline__ void cp_wait0() {
#if HAVE_ASYNC
  __builtin_amdgcn_s_wait_asynccnt(0);
#endif
}

__device__ __forceinline__ v16h make_frag(const h16* p0, const h16* p1) {
  v8h a = *(const v8h*)p0;
  v8h b = *(const v8h*)p1;
  return __builtin_shufflevector(a, b, 0,1,2,3,4,5,6,7,8,9,10,11,12,13,14,15);
}

__device__ __forceinline__ v8f wmma16(v16h a, v16h b, v8f c) {
  return __builtin_amdgcn_wmma_f32_16x16x32_f16(false, a, false, b, (short)0, c,
                                                false, false);
}

__device__ __forceinline__ uint32_t pack2(float a, float b) {
  union { h16 h[2]; uint32_t u; } x;
  x.h[0] = (h16)a; x.h[1] = (h16)b;
  return x.u;
}

// ---------------- fp32 -> f16 weight convert ----------------
__global__ void k_cvt(const float* __restrict__ src, h16* __restrict__ dst, int n) {
  int i = blockIdx.x * blockDim.x + threadIdx.x;
  if (i < n) dst[i] = (h16)src[i];
}

// ---------------- x [B][C][N] -> xT f16 [B][N][C] ----------------
__global__ __launch_bounds__(256) void k_xpose(const float* __restrict__ x,
                                               h16* __restrict__ xT) {
  __shared__ h16 tile[32][33];
  int b = blockIdx.z, c0 = blockIdx.y * 32, n0 = blockIdx.x * 32;
  int tx = threadIdx.x, ty = threadIdx.y;
  for (int k = 0; k < 4; k++) {
    int c = c0 + ty + 8 * k;
    tile[ty + 8 * k][tx] = (h16)x[((size_t)b * CDIM + c) * NPOS + n0 + tx];
  }
  __syncthreads();
  for (int k = 0; k < 4; k++) {
    int n = n0 + ty + 8 * k;
    xT[((size_t)b * NPOS + n) * CDIM + c0 + tx] = tile[tx][ty + 8 * k];
  }
}

// ---------------- QKV GEMM (transposed: rows j, cols n) + RoPE ----------------
// outT[j][n] = sum_c qkv_w[j][c] * xT[n][c];  Q/K -> [b,h,n,d], V -> Vt [b,h,d,n]
__global__ __launch_bounds__(256) void k_qkv_rope(
    const h16* __restrict__ xT, const h16* __restrict__ wqkv,
    h16* __restrict__ Q, h16* __restrict__ K, h16* __restrict__ Vt) {
  const int b = blockIdx.z;
  const int j0 = blockIdx.x * 128;
  const int n0 = blockIdx.y * 128;
  const int tid = threadIdx.x;
  const int w = tid >> 5, lane = tid & 31, lo = lane & 15, hi = lane >> 4;
  const int wj = w & 1, wn = w >> 1;

  __shared__ __align__(16) h16 As[2][128 * 40];
  __shared__ __align__(16) h16 Bs[2][128 * 40];

  v8f acc[4][2] = {};

  const h16* gA = wqkv + (size_t)j0 * CDIM;
  const h16* gB = xT + ((size_t)b * NPOS + n0) * CDIM;
  const int crow = tid >> 1;
  const int ccol = (tid & 1) * 16;

  auto issue = [&](int buf, int kk) {
    cp16(&As[buf][crow * 40 + ccol],     gA + (size_t)crow * CDIM + kk + ccol);
    cp16(&As[buf][crow * 40 + ccol + 8], gA + (size_t)crow * CDIM + kk + ccol + 8);
    cp16(&Bs[buf][crow * 40 + ccol],     gB + (size_t)crow * CDIM + kk + ccol);
    cp16(&Bs[buf][crow * 40 + ccol + 8], gB + (size_t)crow * CDIM + kk + ccol + 8);
  };
  auto compute = [&](int cur) {
    v16h bf[2];
    for (int u = 0; u < 2; u++) {
      const h16* p = &Bs[cur][(wn * 32 + 16 * u + lo) * 40 + 8 * hi];
      bf[u] = make_frag(p, p + 16);
    }
    for (int i = 0; i < 4; i++) {
      const h16* pa = &As[cur][(wj * 64 + 16 * i + lo) * 40 + 8 * hi];
      v16h af = make_frag(pa, pa + 16);
      for (int u = 0; u < 2; u++) acc[i][u] = wmma16(af, bf[u], acc[i][u]);
    }
  };

  const int NT = CDIM / 32;  // 24 k-tiles
  issue(0, 0);
  for (int t = 0; t < NT - 1; ++t) {   // branch-free steady state
    issue((t + 1) & 1, (t + 1) * 32);  // prefetch next tile during compute
    cp_wait4();
    __syncthreads();
    compute(t & 1);
    __syncthreads();
  }
  cp_wait0();
  __syncthreads();
  compute((NT - 1) & 1);

  // epilogue: RoPE on Q/K rows, scatter to Q/K/Vt (f16)
  for (int i = 0; i < 4; i++) {
    int jbase = j0 + wj * 64 + 16 * i + 8 * hi;  // row for r=0 (multiple of 8)
    for (int u = 0; u < 2; u++) {
      int n = n0 + wn * 32 + 16 * u + lo;
      float vals[8];
      for (int r = 0; r < 8; r++) vals[r] = acc[i][u][r];
      if (jbase < 2 * CDIM) {  // Q or K -> RoPE
        int head = (jbase % CDIM) / HDIM;
        int d0 = jbase % HDIM;
        bool isq = jbase < CDIM;
        float pos = (d0 < 32) ? (float)(n >> 5) : (float)(n & 31);
        int pbase = (d0 & 31) >> 1;
        for (int r = 0; r < 8; r += 2) {
          int p = pbase + (r >> 1);
          float invf = __expf(-(float)p * LOG10K_OVER_PAIRS);
          float sn, cs;
          __sincosf(pos * invf, &sn, &cs);
          float a = vals[r], bb = vals[r + 1];
          vals[r]     = a * cs - bb * sn;
          vals[r + 1] = bb * cs + a * sn;
        }
        union { h16 h[8]; uint4 u4; } pk;
        for (int r = 0; r < 8; r++) pk.h[r] = (h16)vals[r];
        h16* dst = (isq ? Q : K) +
                   (((size_t)b * NHEADS + head) * NPOS + n) * HDIM + d0;
        *(uint4*)dst = pk.u4;
      } else {  // V -> Vt [b,h,d,n]
        int head = (jbase - 2 * CDIM) / HDIM;
        int d0 = (jbase - 2 * CDIM) % HDIM;
        h16* dst = Vt + (((size_t)b * NHEADS + head) * HDIM + d0) * NPOS + n;
        for (int r = 0; r < 8; r++) dst[(size_t)r * NPOS] = (h16)vals[r];
      }
    }
  }
}

// ---------------- flash attention (transposed S/O) + entropy ----------------
__global__ __launch_bounds__(256) void k_attn(
    const h16* __restrict__ Q, const h16* __restrict__ K,
    const h16* __restrict__ Vt, h16* __restrict__ attn_out,
    float* __restrict__ ent) {
  const int bh = blockIdx.y;  // b*12 + head
  const int b = bh / NHEADS, head = bh % NHEADS;
  const int tid = threadIdx.x;
  const int w = tid >> 5, lane = tid & 31, lo = lane & 15, hi = lane >> 4;
  const int q0 = blockIdx.x * 128 + w * 16;

  // double-buffered 64-key tiles: K rows [key][d], V rows [d][key]
  __shared__ __align__(16) h16 Ks[2][64 * 72];
  __shared__ __align__(16) h16 Vs[2][64 * 72];

  const size_t base = (size_t)bh * NPOS * HDIM;
  const h16* qp = Q + base + (size_t)(q0 + lo) * HDIM;
  v16h qf0 = make_frag(qp + 8 * hi, qp + 8 * hi + 16);
  v16h qf1 = make_frag(qp + 32 + 8 * hi, qp + 32 + 8 * hi + 16);

  v8f o[4] = {};
  float m = -1e30f, Z = 0.f, T = 0.f;

  const int crK = tid >> 2;           // 0..63
  const int ccK = (tid & 3) * 16;     // 0,16,32,48

  auto issueKV = [&](int buf, int kb) {
    const h16* gK = K + base + (size_t)(kb + crK) * HDIM + ccK;
    cp16(&Ks[buf][crK * 72 + ccK],     gK);
    cp16(&Ks[buf][crK * 72 + ccK + 8], gK + 8);
    const h16* gV = Vt + base + (size_t)crK * NPOS + kb + ccK;
    cp16(&Vs[buf][crK * 72 + ccK],     gV);
    cp16(&Vs[buf][crK * 72 + ccK + 8], gV + 8);
  };

  auto compute = [&](int cur) {
    for (int jb = 0; jb < 64; jb += 32) {
      // S^T tiles: c0 = keys [jb,jb+16), c1 = keys [jb+16,jb+32)
      v8f c0 = {}, c1 = {};
      {
        const h16* pk0 = &Ks[cur][(jb + lo) * 72 + 8 * hi];
        c0 = wmma16(make_frag(pk0, pk0 + 16), qf0, c0);
        c0 = wmma16(make_frag(pk0 + 32, pk0 + 48), qf1, c0);
        const h16* pk1 = &Ks[cur][(jb + 16 + lo) * 72 + 8 * hi];
        c1 = wmma16(make_frag(pk1, pk1 + 16), qf0, c1);
        c1 = wmma16(make_frag(pk1 + 32, pk1 + 48), qf1, c1);
      }
      float s0[8], s1[8], mb = -1e30f;
      for (int r = 0; r < 8; r++) {
        s0[r] = c0[r] * ATTN_SCALE;
        s1[r] = c1[r] * ATTN_SCALE;
        mb = fmaxf(mb, fmaxf(s0[r], s1[r]));
      }
      mb = fmaxf(mb, __shfl_xor(mb, 16));
      float nm = fmaxf(m, mb);
      float sc = __expf(m - nm);
      float p0[8], p1[8], zb = 0.f, tb = 0.f;
      for (int r = 0; r < 8; r++) {
        p0[r] = __expf(s0[r] - nm);
        p1[r] = __expf(s1[r] - nm);
        zb += p0[r] + p1[r];
        tb += p0[r] * s0[r] + p1[r] * s1[r];
      }
      Z = Z * sc + zb;
      T = T * sc + tb;
      m = nm;
      for (int i = 0; i < 4; i++) o[i] *= sc;
      // pack P^T into B fragment (col = query lo, K = keys 16*hi..16*hi+15)
      uint32_t c0p[4], c1p[4];
      for (int r = 0; r < 4; r++) {
        c0p[r] = pack2(p0[2 * r], p0[2 * r + 1]);
        c1p[r] = pack2(p1[2 * r], p1[2 * r + 1]);
      }
      union { uint32_t u[8]; v16h v; } bfu;
      for (int r = 0; r < 4; r++) {
        uint32_t t0 = __shfl_xor(c0p[r], 16);
        uint32_t t1 = __shfl_xor(c1p[r], 16);
        bfu.u[r]     = hi ? t1 : c0p[r];
        bfu.u[r + 4] = hi ? c1p[r] : t0;
      }
      // O^T += Vt * P^T
      for (int i = 0; i < 4; i++) {
        const h16* pv = &Vs[cur][(16 * i + lo) * 72 + jb + 16 * hi];
        o[i] = wmma16(make_frag(pv, pv + 8), bfu.v, o[i]);
      }
    }
  };

  const int NT = NPOS / 64;  // 16 key tiles
  issueKV(0, 0);
  for (int t = 0; t < NT - 1; ++t) {      // branch-free steady state
    issueKV((t + 1) & 1, (t + 1) * 64);   // stream next K/V tile during compute
    cp_wait4();
    __syncthreads();
    compute(t & 1);
    __syncthreads();
  }
  cp_wait0();
  __syncthreads();
  compute((NT - 1) & 1);

  Z += __shfl_xor(Z, 16);
  T += __shfl_xor(T, 16);
  float invZ = 1.f / Z;
  float H = m + __logf(Z) - T * invZ;
  if (hi == 0) ent[(size_t)bh * NPOS + q0 + lo] = H;
  for (int i = 0; i < 4; i++) {
    union { h16 h[8]; uint4 u4; } pk;
    for (int r = 0; r < 8; r++) pk.h[r] = (h16)(o[i][r] * invZ);
    h16* dst = attn_out + ((size_t)b * NPOS + q0 + lo) * CDIM + head * HDIM +
               16 * i + 8 * hi;
    *(uint4*)dst = pk.u4;
  }
}

// ---------------- proj GEMM: out[b][c][n] = attn[b][n][:]·proj_w[c][:] + b[c] ---
__global__ __launch_bounds__(256) void k_proj(
    const h16* __restrict__ A, const h16* __restrict__ Wp,
    const float* __restrict__ bias, float* __restrict__ out) {
  const int b = blockIdx.z;
  const int n0 = blockIdx.x * 128;
  const int c0 = blockIdx.y * 128;
  const int tid = threadIdx.x;
  const int w = tid >> 5, lane = tid & 31, lo = lane & 15, hi = lane >> 4;
  const int wn = w & 1, wc = w >> 1;

  __shared__ __align__(16) h16 As[2][128 * 40];
  __shared__ __align__(16) h16 Bs[2][128 * 40];

  v8f acc[4][2] = {};
  const h16* gA = A + ((size_t)b * NPOS + n0) * CDIM;
  const h16* gB = Wp + (size_t)c0 * CDIM;
  const int crow = tid >> 1;
  const int ccol = (tid & 1) * 16;

  auto issue = [&](int buf, int kk) {
    cp16(&As[buf][crow * 40 + ccol],     gA + (size_t)crow * CDIM + kk + ccol);
    cp16(&As[buf][crow * 40 + ccol + 8], gA + (size_t)crow * CDIM + kk + ccol + 8);
    cp16(&Bs[buf][crow * 40 + ccol],     gB + (size_t)crow * CDIM + kk + ccol);
    cp16(&Bs[buf][crow * 40 + ccol + 8], gB + (size_t)crow * CDIM + kk + ccol + 8);
  };
  auto compute = [&](int cur) {
    v16h bf[2];
    for (int u = 0; u < 2; u++) {
      const h16* p = &Bs[cur][(wc * 32 + 16 * u + lo) * 40 + 8 * hi];
      bf[u] = make_frag(p, p + 16);
    }
    for (int i = 0; i < 4; i++) {
      const h16* pa = &As[cur][(wn * 64 + 16 * i + lo) * 40 + 8 * hi];
      v16h af = make_frag(pa, pa + 16);
      for (int u = 0; u < 2; u++) acc[i][u] = wmma16(af, bf[u], acc[i][u]);
    }
  };

  const int NT = CDIM / 32;
  issue(0, 0);
  for (int t = 0; t < NT - 1; ++t) {
    issue((t + 1) & 1, (t + 1) * 32);
    cp_wait4();
    __syncthreads();
    compute(t & 1);
    __syncthreads();
  }
  cp_wait0();
  __syncthreads();
  compute((NT - 1) & 1);

  for (int i = 0; i < 4; i++) {
    int nbase = n0 + wn * 64 + 16 * i + 8 * hi;
    for (int u = 0; u < 2; u++) {
      int c = c0 + wc * 32 + 16 * u + lo;
      float bv = bias[c];
      float4 f0 = make_float4(acc[i][u][0] + bv, acc[i][u][1] + bv,
                              acc[i][u][2] + bv, acc[i][u][3] + bv);
      float4 f1 = make_float4(acc[i][u][4] + bv, acc[i][u][5] + bv,
                              acc[i][u][6] + bv, acc[i][u][7] + bv);
      float* dst = out + ((size_t)b * CDIM + c) * NPOS + nbase;
      *(float4*)dst = f0;
      *(float4*)(dst + 4) = f1;
    }
  }
}

// ---------------- entropy mean over heads ----------------
__global__ void k_ent(const float* __restrict__ ent, float* __restrict__ u) {
  int i = blockIdx.x * blockDim.x + threadIdx.x;
  if (i < BATCH * NPOS) {
    int b = i >> 10, n = i & 1023;
    float s = 0.f;
    for (int h = 0; h < NHEADS; h++)
      s += ent[((size_t)(b * NHEADS + h)) * NPOS + n];
    u[i] = s * (1.f / NHEADS);
  }
}

extern "C" void kernel_launch(void* const* d_in, const int* in_sizes, int n_in,
                              void* d_out, int out_size, void* d_ws,
                              size_t ws_size, hipStream_t stream) {
  (void)in_sizes; (void)n_in; (void)out_size; (void)ws_size;
  const float* x      = (const float*)d_in[0];
  const float* qkv_w  = (const float*)d_in[1];
  const float* proj_w = (const float*)d_in[2];
  const float* proj_b = (const float*)d_in[3];
  float* out = (float*)d_out;

  char* ws = (char*)d_ws;
  size_t off = 0;
  auto alloc = [&](size_t bytes) {
    void* p = ws + off;
    off = (off + bytes + 255) & ~(size_t)255;
    return p;
  };
  h16* xT   = (h16*)alloc((size_t)BATCH * NPOS * CDIM * 2);
  h16* wqh  = (h16*)alloc((size_t)QKV_ROWS * CDIM * 2);
  h16* wph  = (h16*)alloc((size_t)CDIM * CDIM * 2);
  h16* Qb   = (h16*)alloc((size_t)BATCH * NHEADS * NPOS * HDIM * 2);
  h16* Kb   = (h16*)alloc((size_t)BATCH * NHEADS * NPOS * HDIM * 2);
  h16* Vtb  = (h16*)alloc((size_t)BATCH * NHEADS * NPOS * HDIM * 2);
  h16* attn = (h16*)alloc((size_t)BATCH * NPOS * CDIM * 2);
  float* entb = (float*)alloc((size_t)BATCH * NHEADS * NPOS * 4);

  k_cvt<<<(QKV_ROWS * CDIM + 255) / 256, 256, 0, stream>>>(qkv_w, wqh,
                                                           QKV_ROWS * CDIM);
  k_cvt<<<(CDIM * CDIM + 255) / 256, 256, 0, stream>>>(proj_w, wph,
                                                       CDIM * CDIM);
  k_xpose<<<dim3(NPOS / 32, CDIM / 32, BATCH), dim3(32, 8), 0, stream>>>(x, xT);
  k_qkv_rope<<<dim3(QKV_ROWS / 128, NPOS / 128, BATCH), 256, 0, stream>>>(
      xT, wqh, Qb, Kb, Vtb);
  k_attn<<<dim3(NPOS / 128, BATCH * NHEADS), 256, 0, stream>>>(Qb, Kb, Vtb,
                                                               attn, entb);
  k_proj<<<dim3(NPOS / 128, CDIM / 128, BATCH), 256, 0, stream>>>(attn, wph,
                                                                  proj_b, out);
  k_ent<<<(BATCH * NPOS + 255) / 256, 256, 0, stream>>>(
      entb, out + (size_t)BATCH * CDIM * NPOS);
}
